// SGCLayer_40381282517583
// MI455X (gfx1250) — matrix operand order repaired and checked by reading, not compiled, not run on hardware
//
#include <hip/hip_runtime.h>
#include <math.h>

// SGC k=2 on MI455X (gfx1250, wave32).
// Memory-bound gather/segment-sum; feature matrix (25.6MB) is L2-resident (192MB).
// Segment-sum over dst-sorted edge blocks as P(16x32,0/1) x M(32x16) on the WMMA
// pipe (split hi/lo f16 -> f32-accurate), sources staged to LDS with the CDNA5
// async-to-LDS engine (global_load_async_to_lds_b128 + s_wait_asynccnt).

typedef __attribute__((ext_vector_type(16))) _Float16 v16h;
typedef __attribute__((ext_vector_type(8)))  float    v8f;

#define D_FEAT 128
#define EB 32            // edges per WMMA block == K of v_wmma_f32_16x16x32_f16

#if defined(__has_builtin)
#if __has_builtin(__builtin_amdgcn_global_load_async_to_lds_b128) && \
    __has_builtin(__builtin_amdgcn_s_wait_asynccnt)
#define USE_ASYNC_LDS 1
#endif
#endif

// builtin prototype (from clang diagnostic): (v4i AS1*, v4i AS3*, Imm int, Imm int)
typedef int v4i_gcc __attribute__((vector_size(16)));
typedef __attribute__((address_space(1))) v4i_gcc GAS_v4i;
typedef __attribute__((address_space(3))) v4i_gcc LAS_v4i;

// ---------------- small helper kernels ----------------

__global__ void deg_kernel(const int* __restrict__ dst, unsigned* __restrict__ deg, int E) {
  int e = blockIdx.x * blockDim.x + threadIdx.x;
  if (e < E) atomicAdd(&deg[dst[e]], 1u);
}

__global__ void norm_kernel(const unsigned* __restrict__ deg, float* __restrict__ nrm, int N) {
  int v = blockIdx.x * blockDim.x + threadIdx.x;
  if (v < N) nrm[v] = rsqrtf(fmaxf((float)deg[v], 1.0f));
}

// one-workgroup blocked exclusive scan of deg -> rowptr
__global__ void scan_kernel(const unsigned* __restrict__ deg, unsigned* __restrict__ rowptr, int N) {
  __shared__ unsigned part[1024];
  int t = threadIdx.x;
  int chunk = (N + 1023) >> 10;
  int s = t * chunk;
  int e = min(s + chunk, N);
  unsigned sum = 0;
  for (int i = s; i < e; ++i) sum += deg[i];
  part[t] = sum;
  __syncthreads();
  for (int off = 1; off < 1024; off <<= 1) {
    unsigned v = (t >= off) ? part[t - off] : 0u;
    __syncthreads();
    part[t] += v;
    __syncthreads();
  }
  unsigned run = (t == 0) ? 0u : part[t - 1];
  for (int i = s; i < e; ++i) { rowptr[i] = run; run += deg[i]; }
  if (t == 1023) rowptr[N] = part[1023];
}

// counting-sort edges by dst into (ssrc, sdst)
__global__ void scatter_kernel(const int* __restrict__ src, const int* __restrict__ dst,
                               const unsigned* __restrict__ rowptr, unsigned* __restrict__ cursor,
                               int* __restrict__ ssrc, int* __restrict__ sdst, int E) {
  int e = blockIdx.x * blockDim.x + threadIdx.x;
  if (e < E) {
    int d = dst[e];
    unsigned p = rowptr[d] + atomicAdd(&cursor[d], 1u);
    ssrc[p] = src[e];
    sdst[p] = d;
  }
}

__global__ void scale_kernel(float* __restrict__ h, const float* __restrict__ nrm, int total) {
  int i = blockIdx.x * blockDim.x + threadIdx.x;
  if (i < total) h[i] *= nrm[i >> 7];
}

// out = (features + h1_scaled + raw2 * norm[v]) / 3   (raw2 lives in `out`)
__global__ void final_kernel(const float* __restrict__ f, const float* __restrict__ h1,
                             const float* __restrict__ nrm, float* __restrict__ out, int total) {
  int i = blockIdx.x * blockDim.x + threadIdx.x;
  if (i < total) out[i] = (f[i] + h1[i] + out[i] * nrm[i >> 7]) * (1.0f / 3.0f);
}

// ---------------- WMMA segment-sum hop kernel ----------------
// One wave32 per block of 32 dst-sorted edges:
//   stage 32 source rows (16KB) to LDS via async-to-LDS b128 (coalesced 512B/instr)
//   P (16x32 f16): P[m][k] = 1 if edge k belongs to the m-th distinct dst run
//   M split into hi/lo f16; D(16x16 f32) = P*hi + P*lo  (f32-accurate)
//   distinct rows scattered with global_atomic_add_f32 (cross-block runs ok)
__global__ void __launch_bounds__(128)
hop_wmma_kernel(const float* __restrict__ hin, const int* __restrict__ ssrc,
                const int* __restrict__ sdst, const float* __restrict__ nrm,
                float* __restrict__ hout, int nEB, int E) {
  __shared__ float stage[4 * EB * D_FEAT];          // 64KB: 16KB per wave
  const int lane = threadIdx.x & 31;
  const int wv   = threadIdx.x >> 5;
  const int eb   = blockIdx.x * 4 + wv;
  if (eb >= nEB) return;                            // wave-uniform
  float* st = &stage[wv * EB * D_FEAT];
  const int base = eb * EB;
  const int nval = min(EB, E - base);

  // per-lane edge data
  const int  e     = base + lane;
  const bool valid = (e < E);
  const int  d_own = valid ? sdst[e] : -1;
  const int  s_own = valid ? ssrc[e] : 0;
  const float w_own = valid ? nrm[s_own] : 0.0f;
  const int  d_prev = (valid && lane > 0) ? sdst[e - 1] : -1;
  const bool isStart = valid && (lane == 0 || d_prev != d_own);
  const unsigned mask = (unsigned)__ballot(isStart);
  const int ndist = __popc(mask);                   // #distinct dsts in block

  // ---- stage the 32 source rows into LDS (one row per iteration, 512B coalesced)
#if USE_ASYNC_LDS
  for (int i = 0; i < EB; ++i) {
    int s_i = __shfl(s_own, i, 32);
    const float* g = hin + (size_t)s_i * D_FEAT + lane * 4;
    float* l = st + i * D_FEAT + lane * 4;
    __builtin_amdgcn_global_load_async_to_lds_b128((GAS_v4i*)g, (LAS_v4i*)l, 0, 0);
  }
  __builtin_amdgcn_s_wait_asynccnt(0);
#else
  for (int i = 0; i < EB; ++i) {
    int s_i = __shfl(s_own, i, 32);
    const float4 v = *(const float4*)(hin + (size_t)s_i * D_FEAT + lane * 4);
    *(float4*)(st + i * D_FEAT + lane * 4) = v;
  }
#endif

  // Build A = P, f16 16x32, contiguous k order matching B:
  // lanes<16: K0..15 ; lanes>=16: K16..31
  const int m = lane & 15;
  v16h a;
#pragma unroll
  for (int j = 0; j < 16; ++j) {
    int K   = j + ((lane < 16) ? 0 : 16);
    int row = __popc(mask & (0xFFFFFFFFu >> (31 - K))) - 1;   // run index of edge K
    a[j] = (_Float16)((K < nval && row == m) ? 1.0f : 0.0f);
  }

  // per-K weights for B building (B rows this lane touches: K = Kbase..Kbase+15)
  const int Kbase = (lane < 16) ? 0 : 16;
  float weK[16];
#pragma unroll
  for (int j = 0; j < 16; ++j) weK[j] = __shfl(w_own, Kbase + j, 32);

  // node id for each of the up-to-16 output rows (computed with all lanes active)
  int rowNode[8];
#pragma unroll
  for (int r = 0; r < 8; ++r) {
    int mm = r + ((lane < 16) ? 0 : 8);
    unsigned tm = mask;
    for (int i = 0; i < mm; ++i) tm &= tm - 1;   // clear mm lowest set bits
    int sl = __ffs(tm) - 1;
    int nd = __shfl(d_own, (sl < 0) ? 0 : sl, 32);
    rowNode[r] = (mm < ndist) ? nd : -1;
  }

  const int n = lane & 15;
#pragma unroll 2
  for (int ng = 0; ng < 8; ++ng) {             // 8 x 16 feature columns
    const int ncb = ng * 16;
    v16h bh, bl;                               // hi/lo split of messages
#pragma unroll
    for (int j = 0; j < 16; ++j) {
      float mv = weK[j] * st[(Kbase + j) * D_FEAT + ncb + n];
      _Float16 hi = (_Float16)mv;
      bh[j] = hi;
      bl[j] = (_Float16)(mv - (float)hi);
    }

    v8f c = {};
    c = __builtin_amdgcn_wmma_f32_16x16x32_f16(false, a, false, bh, (short)0, c, false, false);
    c = __builtin_amdgcn_wmma_f32_16x16x32_f16(false, a, false, bl, (short)0, c, false, false);

    // D layout: VGPR r -> (M=r, N=lane) lanes<16 ; (M=r+8, N=lane-16) lanes>=16
#pragma unroll
    for (int r = 0; r < 8; ++r) {
      int node = rowNode[r];
      if (node >= 0)
        atomicAdd(&hout[(size_t)node * D_FEAT + ncb + n], c[r]);
    }
  }
}

// ---------------- launch ----------------

extern "C" void kernel_launch(void* const* d_in, const int* in_sizes, int n_in,
                              void* d_out, int out_size, void* d_ws, size_t ws_size,
                              hipStream_t stream) {
  const float* feat = (const float*)d_in[0];
  const int*   src  = (const int*)d_in[1];   // jax default: int64 demoted to int32
  const int*   dst  = (const int*)d_in[2];
  float* out = (float*)d_out;
  const int E = in_sizes[1];
  const int N = in_sizes[0] / D_FEAT;
  const int total = N * D_FEAT;

  char* ws = (char*)d_ws;
  size_t off = 0;
  auto take = [&](size_t bytes) -> void* {
    void* p = ws + off;
    off = (off + bytes + 255) & ~(size_t)255;
    return p;
  };
  unsigned* deg    = (unsigned*)take((size_t)N * 4);
  unsigned* cursor = (unsigned*)take((size_t)N * 4);
  unsigned* rowptr = (unsigned*)take((size_t)(N + 1) * 4);
  float*    nrm    = (float*)take((size_t)N * 4);
  int*      ssrc   = (int*)take((size_t)E * 4);
  int*      sdst   = (int*)take((size_t)E * 4);
  float*    h1     = (float*)take((size_t)total * 4);
  (void)ws_size; (void)n_in; (void)out_size;

  (void)hipMemsetAsync(deg, 0, (size_t)N * 4, stream);
  (void)hipMemsetAsync(cursor, 0, (size_t)N * 4, stream);
  (void)hipMemsetAsync(h1, 0, (size_t)total * 4, stream);

  deg_kernel   <<<(E + 255) / 256, 256, 0, stream>>>(dst, deg, E);
  norm_kernel  <<<(N + 255) / 256, 256, 0, stream>>>(deg, nrm, N);
  scan_kernel  <<<1, 1024, 0, stream>>>(deg, rowptr, N);
  scatter_kernel<<<(E + 255) / 256, 256, 0, stream>>>(src, dst, rowptr, cursor, ssrc, sdst, E);

  const int nEB = (E + EB - 1) / EB;
  // hop 1: raw1 = A_hat_unscaled @ features  (weight norm[src] folded in)
  hop_wmma_kernel<<<(nEB + 3) / 4, 128, 0, stream>>>(feat, ssrc, sdst, nrm, h1, nEB, E);
  // h1 = raw1 * norm[dst]
  scale_kernel<<<(total + 255) / 256, 256, 0, stream>>>(h1, nrm, total);
  // hop 2 accumulates into d_out (zeroed first)
  (void)hipMemsetAsync(out, 0, (size_t)total * 4, stream);
  hop_wmma_kernel<<<(nEB + 3) / 4, 128, 0, stream>>>(h1, ssrc, sdst, nrm, out, nEB, E);
  // out = (features + h1 + out*norm) / 3
  final_kernel<<<(total + 255) / 256, 256, 0, stream>>>(feat, h1, nrm, out, total);
}
